// TernaryMobiusLinear_87625922773407
// MI455X (gfx1250) — matrix operand order repaired and checked by reading, not compiled, not run on hardware
//
#include <hip/hip_runtime.h>
#include <hip/hip_bf16.h>
#include <math.h>

// Problem constants (from reference): B=4, L=2048, N=2048, K=2048
#define TM_TOTAL 8192   // B*L rows
#define TN       2048   // reduction dim
#define TK       2048   // output cols
#define BM       128    // block tile rows   (4 M-waves * 32)
#define BK       128    // block tile cols   (2 C-waves * 64)
#define KC       32     // reduction chunk (WMMA bf16 depth)
#define LDA      40     // padded LDS row stride in bf16 elements (conflict-free)
#define NCHUNK   (TN / KC)

typedef __attribute__((ext_vector_type(16))) __bf16 v16bf;
typedef __attribute__((ext_vector_type(8)))  float  v8f;

union FragU { uint4 q[2]; v16bf v; };
union Pack8 { __bf16 b[8]; uint4 q; };

// Stage one K-chunk: x(128x32 f32) -> hi/lo bf16 LDS tiles, w(32x128 f32) -> transposed bf16 LDS tile.
__device__ __forceinline__ void stage_tile(const float* __restrict__ x,
                                           const float* __restrict__ w,
                                           __bf16* __restrict__ sAh,
                                           __bf16* __restrict__ sAl,
                                           __bf16* __restrict__ sBt,
                                           int m0, int kc0, int n0, int tid)
{
    // ---- A: 128 rows x 32 cols, 16 contiguous floats per thread ----
    const int ar = tid >> 1;          // 0..127
    const int ac = (tid & 1) << 4;    // 0 or 16
    const float* xp = x + (size_t)(m0 + ar) * TN + n0 + ac;
    float4 f0 = *(const float4*)(xp);
    float4 f1 = *(const float4*)(xp + 4);
    float4 f2 = *(const float4*)(xp + 8);
    float4 f3 = *(const float4*)(xp + 12);
    float fv[16] = {f0.x, f0.y, f0.z, f0.w, f1.x, f1.y, f1.z, f1.w,
                    f2.x, f2.y, f2.z, f2.w, f3.x, f3.y, f3.z, f3.w};
    Pack8 ph[2], pl[2];
#pragma unroll
    for (int i = 0; i < 16; ++i) {
        __bf16 h = (__bf16)fv[i];
        ph[i >> 3].b[i & 7] = h;
        pl[i >> 3].b[i & 7] = (__bf16)(fv[i] - (float)h);  // residual: ~8 extra mantissa bits
    }
    *(uint4*)(sAh + ar * LDA + ac)     = ph[0].q;
    *(uint4*)(sAh + ar * LDA + ac + 8) = ph[1].q;
    *(uint4*)(sAl + ar * LDA + ac)     = pl[0].q;
    *(uint4*)(sAl + ar * LDA + ac + 8) = pl[1].q;

    // ---- B: 32 rows x 128 cols, stored transposed [col][k] ----
#pragma unroll
    for (int u = 0; u < 4; ++u) {
        int unit = tid + u * 256;        // 0..1023 float4 units
        int kk   = unit >> 5;            // 0..31 (reduction row)
        int cq   = (unit & 31) << 2;     // 0..124 (col quad)
        const float* wp = w + (size_t)(n0 + kk) * TK + kc0 + cq;
        float4 f = *(const float4*)wp;
        sBt[(cq + 0) * LDA + kk] = (__bf16)f.x;
        sBt[(cq + 1) * LDA + kk] = (__bf16)f.y;
        sBt[(cq + 2) * LDA + kk] = (__bf16)f.z;
        sBt[(cq + 3) * LDA + kk] = (__bf16)f.w;
    }
}

// GEMM: out[m,k] = sum_n x[m,n]*w[n,k] * scale[k], split-bf16 (hi+lo) via WMMA.
__global__ __launch_bounds__(256)
void ternary_gemm_wmma(const float* __restrict__ x,
                       const float* __restrict__ w,
                       const float* __restrict__ scale,
                       float* __restrict__ out)
{
    // double-buffered LDS: 2 * (10.0 + 10.0 + 10.0) KB = 60 KB
    __shared__ __bf16 sAh[2][BM * LDA];
    __shared__ __bf16 sAl[2][BM * LDA];
    __shared__ __bf16 sBt[2][BK * LDA];

    const int tid  = threadIdx.x;
    const int lane = tid & 31;
    const int wave = tid >> 5;     // 0..7
    const int half = lane >> 4;    // 0/1 (lane group)
    const int l16  = lane & 15;
    const int wm   = wave >> 1;    // 0..3 -> 32-row slice
    const int wc   = wave & 1;     // 0..1 -> 64-col slice

    const int m0  = blockIdx.y * BM;
    const int kc0 = blockIdx.x * BK;

    const v8f vzero = {0.f, 0.f, 0.f, 0.f, 0.f, 0.f, 0.f, 0.f};
    v8f acc[2][4];
#pragma unroll
    for (int i = 0; i < 2; ++i)
#pragma unroll
        for (int j = 0; j < 4; ++j) acc[i][j] = vzero;

    // prologue: stage chunk 0 into buffer 0
    stage_tile(x, w, sAh[0], sAl[0], sBt[0], m0, kc0, 0, tid);
    __syncthreads();

    for (int c = 0; c < NCHUNK; ++c) {
        const int cur = c & 1;

        // stage next chunk into the other buffer (overlaps with WMMA below)
        if (c + 1 < NCHUNK)
            stage_tile(x, w, sAh[cur ^ 1], sAl[cur ^ 1], sBt[cur ^ 1],
                       m0, kc0, (c + 1) * KC, tid);

        const __bf16* Ah = sAh[cur];
        const __bf16* Al = sAl[cur];
        const __bf16* Bt = sBt[cur];

        // B fragments (32x16 bf16): lane l -> col l%16; elem e<8 -> K = e + 8*half; e>=8 -> K = e + 8 + 8*half
        v16bf bbf[4];
#pragma unroll
        for (int j = 0; j < 4; ++j) {
            const __bf16* cp = Bt + (wc * 64 + j * 16 + l16) * LDA;
            FragU u;
            u.q[0] = *(const uint4*)(cp + 8 * half);
            u.q[1] = *(const uint4*)(cp + 16 + 8 * half);
            bbf[j] = u.v;
        }

        // A fragments (16x32 bf16): lane l -> row l%16;
        //   elems e<8 : K = e + 8*half ; elems e>=8 : K = e + 8 + 8*half
#pragma unroll
        for (int i = 0; i < 2; ++i) {
            const __bf16* rp = Ah + (wm * 32 + i * 16 + l16) * LDA;
            FragU u0;
            u0.q[0] = *(const uint4*)(rp + 8 * half);
            u0.q[1] = *(const uint4*)(rp + 16 + 8 * half);
            v16bf ah = u0.v;
            const __bf16* rq = Al + (wm * 32 + i * 16 + l16) * LDA;
            FragU u1;
            u1.q[0] = *(const uint4*)(rq + 8 * half);
            u1.q[1] = *(const uint4*)(rq + 16 + 8 * half);
            v16bf al = u1.v;

            // hi pass then lo pass: each accumulator reused 4 WMMAs apart
#pragma unroll
            for (int j = 0; j < 4; ++j)
                acc[i][j] = __builtin_amdgcn_wmma_f32_16x16x32_bf16(
                    false, ah, false, bbf[j], (short)0, acc[i][j], false, false);
#pragma unroll
            for (int j = 0; j < 4; ++j)
                acc[i][j] = __builtin_amdgcn_wmma_f32_16x16x32_bf16(
                    false, al, false, bbf[j], (short)0, acc[i][j], false, false);
        }

        __syncthreads();
    }

    // ---- epilogue: per-column scale, write raw result to d_out ----
    // C/D layout: VGPR r, lane l -> row = r + 8*half, col = l%16
#pragma unroll
    for (int j = 0; j < 4; ++j) {
        int   col = kc0 + wc * 64 + j * 16 + l16;
        float s   = scale[col];
#pragma unroll
        for (int i = 0; i < 2; ++i) {
            int    row = m0 + wm * 32 + i * 16 + half * 8;
            float* op  = out + (size_t)row * TK + col;
#pragma unroll
            for (int r = 0; r < 8; ++r)
                op[(size_t)r * TK] = acc[i][j][r] * s;
        }
    }
}

// Pass 2: per-row Poincare exp-map + ball clamp, in place on d_out.
__global__ __launch_bounds__(256)
void poincare_rownorm(float* __restrict__ out)
{
    const float eps = 1e-7f;
    const int   row = blockIdx.x;
    float*      p   = out + (size_t)row * TK;
    const int   tid = threadIdx.x;

    float4 v0 = *(float4*)(p + tid * 8);
    float4 v1 = *(float4*)(p + tid * 8 + 4);
    float  ss = v0.x * v0.x + v0.y * v0.y + v0.z * v0.z + v0.w * v0.w
              + v1.x * v1.x + v1.y * v1.y + v1.z * v1.z + v1.w * v1.w;

    // wave32 butterfly reduction
#pragma unroll
    for (int off = 16; off > 0; off >>= 1)
        ss += __shfl_xor(ss, off, 32);

    __shared__ float red[8];
    if ((tid & 31) == 0) red[tid >> 5] = ss;
    __syncthreads();
    float total = red[0] + red[1] + red[2] + red[3]
                + red[4] + red[5] + red[6] + red[7];

    float norm = sqrtf(total);
    float vn   = fmaxf(norm, eps);              // v_norm (sqrt_c = 1)
    float t    = tanhf(vn) / (vn + eps);        // exp-map factor
    float on   = fmaxf(norm * t, eps);          // norm after exp-map
    float f    = t * fminf(0.99f / on, 1.0f);   // + ball clamp

    v0.x *= f; v0.y *= f; v0.z *= f; v0.w *= f;
    v1.x *= f; v1.y *= f; v1.z *= f; v1.w *= f;
    *(float4*)(p + tid * 8)     = v0;
    *(float4*)(p + tid * 8 + 4) = v1;
}

extern "C" void kernel_launch(void* const* d_in, const int* in_sizes, int n_in,
                              void* d_out, int out_size, void* d_ws, size_t ws_size,
                              hipStream_t stream)
{
    const float* x     = (const float*)d_in[0];   // (4, 2048, 2048) f32
    const float* w     = (const float*)d_in[1];   // (2048, 2048) ternary f32
    const float* scale = (const float*)d_in[2];   // (2048,) f32
    float*       out   = (float*)d_out;           // (4, 2048, 2048) f32

    dim3 grid(TK / BK, TM_TOTAL / BM);            // (16, 64)
    ternary_gemm_wmma<<<grid, 256, 0, stream>>>(x, w, scale, out);
    poincare_rownorm<<<TM_TOTAL, 256, 0, stream>>>(out);
}
